// CostVolumeLayer_27187142983991
// MI455X (gfx1250) — compile-verified, matrix operand order checked by reading, not run on hardware
//
#include <hip/hip_runtime.h>

typedef __attribute__((ext_vector_type(16))) _Float16 v16h;
typedef __attribute__((ext_vector_type(8)))  float    v8f;
typedef __attribute__((ext_vector_type(4)))  float    v4f;

#define B_      4
#define H_      128
#define W_      128
#define C_      128
#define BLK     4
#define S_      15
#define HC      32
#define WC      32
#define PAD     7          // (S_-1)/2
#define FRDIM   125        // H_-BLK+1 : valid f_r extent
#define FRP     139        // FRDIM + S_-1 : padded f_r extent
#define NSHIFT  225        // S_*S_

#define FC_ELEMS (B_*HC*WC*C_)       // 524288
#define FR_ELEMS (B_*FRP*FRP*C_)     // 9892352
#define FR_OFF   ((size_t)FC_ELEMS*sizeof(_Float16))  // 1 MiB, 256B aligned

union AV { v16h h; unsigned int u[8]; };

// Load one 32x16 f16 B fragment set (4 K-chunks) from a patch channel vector.
// Per-lane K pattern: lanes 0-15 hold K {0-7,16-23}, lanes 16-31 K {8-15,24-31}.
__device__ __forceinline__ void load_b(AV (&bb)[4], const unsigned int* __restrict__ frv,
                                       int kb2) {
    #pragma unroll
    for (int kc = 0; kc < 4; ++kc) {
        const int base = kc * 16 + kb2;
        #pragma unroll
        for (int t = 0; t < 4; ++t) bb[kc].u[t]     = frv[base + t];
        #pragma unroll
        for (int t = 0; t < 4; ++t) bb[kc].u[4 + t] = frv[base + 8 + t];
    }
}

// Dot over C=128 via 4 chained WMMAs; rows of A identical -> acc[0] is this
// lane's N-column result on every lane.
__device__ __forceinline__ float tile_dot(const AV (&a)[4], const AV (&bb)[4]) {
    v8f acc = {};
    #pragma unroll
    for (int kc = 0; kc < 4; ++kc)
        acc = __builtin_amdgcn_wmma_f32_16x16x32_f16(
                  false, a[kc].h, false, bb[kc].h, (short)0, acc, false, false);
    return acc[0];
}

// ---------------------------------------------------------------------------
// Kernel 1: f_c = avg_pool(features_0, k=4, stride=4) -> f16, [b][i][j][c].
// features_0 is read exactly once -> non-temporal loads.
// ---------------------------------------------------------------------------
__global__ void cv_pool_fc(const float* __restrict__ in, _Float16* __restrict__ fc) {
    int idx = blockIdx.x * blockDim.x + threadIdx.x;
    if (idx >= FC_ELEMS) return;
    int c =  idx         & (C_ - 1);
    int j = (idx >> 7)   & (WC - 1);
    int i = (idx >> 12)  & (HC - 1);
    int b =  idx >> 17;
    const float* p = in + (((long)(b * H_ + i * BLK)) * W_ + j * BLK) * C_ + c;
    float s = 0.f;
    #pragma unroll
    for (int ky = 0; ky < BLK; ++ky)
        #pragma unroll
        for (int kx = 0; kx < BLK; ++kx)
            s += __builtin_nontemporal_load(p + (ky * W_ + kx) * C_);
    fc[idx] = (_Float16)(s * (1.0f / 16.0f));
}

// ---------------------------------------------------------------------------
// Kernel 2: f_r_pad = zero-pad(avg_pool(features_0from1, k=4, stride=1)) -> f16
//           [b][y][x][c], y,x in [0,139). Overlapping windows re-read each
//           input element 16x -> keep cached (RT).
// ---------------------------------------------------------------------------
__global__ void cv_pool_fr(const float* __restrict__ in, _Float16* __restrict__ fr) {
    long idx = (long)blockIdx.x * blockDim.x + threadIdx.x;
    if (idx >= FR_ELEMS) return;
    int  c = (int)(idx & (C_ - 1));
    long t = idx >> 7;
    int  x = (int)(t % FRP); t /= FRP;
    int  y = (int)(t % FRP);
    int  b = (int)(t / FRP);
    int y0 = y - PAD, x0 = x - PAD;
    float s = 0.f;
    if (y0 >= 0 && y0 < FRDIM && x0 >= 0 && x0 < FRDIM) {
        const float* p = in + (((long)(b * H_ + y0)) * W_ + x0) * C_ + c;
        #pragma unroll
        for (int ky = 0; ky < BLK; ++ky)
            #pragma unroll
            for (int kx = 0; kx < BLK; ++kx)
                s += p[(ky * W_ + kx) * C_];
        s *= (1.0f / 16.0f);
    }
    fr[idx] = (_Float16)s;
}

// ---------------------------------------------------------------------------
// Kernel 3: correlation via WMMA, one wave32 per coarse cell (b,i,j).
//   Double-buffered B fragments: tile nt+1's 8 global_load_b128 are issued
//   before tile nt's WMMAs so the matrix pipe overlaps L2 latency.
//   Per-tile addressing is incremental: n += 16 <=> (u,v) += (1,1) with wrap
//   at v==14 (off += 264*C instead of 140*C); tile 14's out-of-range shifts
//   clamp to the last valid patch (n=224) via one umin.
// ---------------------------------------------------------------------------
__global__ __launch_bounds__(256) void cv_corr_wmma(const _Float16* __restrict__ fc,
                                                    const _Float16* __restrict__ fr,
                                                    float* __restrict__ out) {
    __shared__ float lds[8 * 240];

    const int lane = threadIdx.x & 31;
    const int wave = threadIdx.x >> 5;
    const int cell = blockIdx.x * 8 + wave;            // 4096 cells total
    const int b =  cell >> 10;
    const int i = (cell >> 5) & 31;
    const int j =  cell & 31;

    const int kb2 = ((lane < 16) ? 0 : 8) >> 1;        // f16-pair (uint) index base
    const int col = lane & 15;                          // N column this lane owns

    // A fragments: f_c vector for this cell, same data in every M row.
    const unsigned int* fcv = (const unsigned int*)(fc + (long)cell * C_);
    AV a[4];
    #pragma unroll
    for (int kc = 0; kc < 4; ++kc) {
        const int base = kc * 16 + kb2;
        #pragma unroll
        for (int t = 0; t < 4; ++t) a[kc].u[t]     = fcv[base + t];
        #pragma unroll
        for (int t = 0; t < 4; ++t) a[kc].u[4 + t] = fcv[base + 8 + t];
    }

    float* wlds = lds + wave * 240;

    // Wave-uniform patch base; per-lane 32-bit element offset (u*FRP+v)*C_.
    const _Float16* frbase = fr + (((long)b * FRP + i * BLK) * FRP + j * BLK) * C_;
    unsigned off = ((col == 15) ? (unsigned)FRP : (unsigned)col) * C_;  // n0=col -> (u0,v0)
    int v = (col == 15) ? 0 : col;

    AV b0[4], b1[4];
    load_b(b0, (const unsigned int*)(frbase + off), kb2);

    #pragma unroll
    for (int nt = 0; nt < 15; ++nt) {
        if (nt + 1 < 15) {
            // advance n by 16: (u,v) -> (u+1, v+1), wrap v at 15
            const unsigned step = (v == 14) ? (unsigned)(264 * C_) : (unsigned)(140 * C_);
            off += step;
            v = (v == 14) ? 0 : v + 1;
            if (nt + 1 == 14 && off > 1960u * C_) off = 1960u * C_;  // clamp to n=224
            if (nt & 1) load_b(b0, (const unsigned int*)(frbase + off), kb2);
            else        load_b(b1, (const unsigned int*)(frbase + off), kb2);
        }
        const float d = (nt & 1) ? tile_dot(a, b1) : tile_dot(a, b0);
        float val = d * (1.0f / (float)C_);
        val = (val > 0.f) ? val : 0.1f * val;           // leaky_relu(0.1)
        int n = nt * 16 + col;
        if (n > NSHIFT - 1) n = NSHIFT - 1;
        wlds[n] = val;                                   // all lanes, no EXEC branch
    }

    __syncthreads();

    // 4x4 replication: each fine pixel gets the cell's 225 shifts, contiguous
    // in the (B,H,W,225) output. 56 x b128 NT stores + 1 tail per pixel.
    const v4f* src = (const v4f*)wlds;                  // 16B-aligned (wave*960B)
    for (int p = 0; p < 16; ++p) {
        const int h = i * BLK + (p >> 2);
        const int w = j * BLK + (p & 3);
        float* op = out + (((long)b * H_ + h) * W_ + w) * NSHIFT;
        for (int t = lane; t < 56; t += 32)
            __builtin_nontemporal_store(src[t], (v4f*)(op + t * 4));
        if (lane == 0)
            __builtin_nontemporal_store(wlds[224], op + 224);
    }
}

// ---------------------------------------------------------------------------
extern "C" void kernel_launch(void* const* d_in, const int* in_sizes, int n_in,
                              void* d_out, int out_size, void* d_ws, size_t ws_size,
                              hipStream_t stream) {
    const float* f0  = (const float*)d_in[0];   // features_0
    const float* f1  = (const float*)d_in[1];   // features_0from1
    float*       out = (float*)d_out;

    _Float16* fcw = (_Float16*)d_ws;
    _Float16* frw = (_Float16*)((char*)d_ws + FR_OFF);

    cv_pool_fc<<<(FC_ELEMS + 255) / 256, 256, 0, stream>>>(f0, fcw);
    cv_pool_fr<<<(int)((FR_ELEMS + 255) / 256), 256, 0, stream>>>(f1, frw);
    cv_corr_wmma<<<(B_ * HC * WC) / 8, 256, 0, stream>>>(fcw, frw, out);
}
